// UncertaintyNNUE_73057393705757
// MI455X (gfx1250) — matrix operand order repaired and checked by reading, not compiled, not run on hardware
//
#include <hip/hip_runtime.h>
#include <hip/hip_bf16.h>
#include <math.h>

// ---------------------------------------------------------------------------
// UncertaintyNNUE on MI455X (gfx1250, wave32, WMMA).
//
// Dominant cost: 2x GEMM (8192x22528)@(22528x128) over 1.48GB fp32 features.
// HBM floor @23.3TB/s ~= 64us -> memory bound with bf16 WMMA.
// Features: stream f32 -> v_cvt_pk_bf16_f32 -> A fragments (read-once data).
// Weights: pre-converted once to bf16 in workspace (L2-resident, 11.5MB),
// B fragments are direct b128 loads with zero per-iteration conversion.
// Wave tile 16x64 (4 WMMAs per A fragment) to amortize the A load+convert.
// ---------------------------------------------------------------------------

#define HD    22528   // HALF_DIMENSIONS
#define L1    128
#define BATCH 8192
#define VH    32
#define NQ    5
#define PO    128

typedef __attribute__((ext_vector_type(16))) __bf16 v16bf;
typedef __attribute__((ext_vector_type(8)))  __bf16 v8bf;
typedef __attribute__((ext_vector_type(8)))  float  v8f;
typedef __attribute__((ext_vector_type(4)))  float  v4f;

__device__ static inline v16bf cvt16(v4f x0, v4f x1, v4f x2, v4f x3) {
    v16bf r;
#pragma unroll
    for (int i = 0; i < 4; ++i) {
        r[i]      = (__bf16)x0[i];
        r[4 + i]  = (__bf16)x1[i];
        r[8 + i]  = (__bf16)x2[i];
        r[12 + i] = (__bf16)x3[i];
    }
    return r;
}

__device__ static inline v8f wmma_bf16(v16bf a, v16bf b, v8f c) {
    // (neg_a, A, neg_b, B, c_mod, C, reuse_a, reuse_b)
    return __builtin_amdgcn_wmma_f32_16x16x32_bf16(false, a, false, b,
                                                   (short)0, c, false, false);
}

// ---------------------------------------------------------------------------
// Prepass: f32 -> bf16 elementwise (n multiple of 8). Packed b128 stores.
// ---------------------------------------------------------------------------
__global__ __launch_bounds__(256) void cvt_weights(
    const float* __restrict__ src, __bf16* __restrict__ dst, int n)
{
    int i = (blockIdx.x * blockDim.x + threadIdx.x) * 8;
    if (i >= n) return;
    v4f a = *(const v4f*)(src + i);
    v4f b = *(const v4f*)(src + i + 4);
    v8bf r;
#pragma unroll
    for (int j = 0; j < 4; ++j) { r[j] = (__bf16)a[j]; r[4 + j] = (__bf16)b[j]; }
    *(v8bf*)(dst + i) = r;
}

// ---------------------------------------------------------------------------
// Kernel 1: feature transform  acc_side = relu(feats @ W^T + b)
// block = 128 threads (4 waves). Wave: 16 batch rows x 64 L1 cols (4 WMMA
// tiles sharing one A fragment). Block: 32 rows x 128 cols.
// grid = (BATCH/32, 2 sides).
// ---------------------------------------------------------------------------
__global__ __launch_bounds__(128) void ft_gemm(
    const float* __restrict__ feats_w, const float* __restrict__ feats_b,
    const __bf16* __restrict__ Wwbf, const float* __restrict__ bw,
    const __bf16* __restrict__ Wbbf, const float* __restrict__ bb,
    float* __restrict__ acc_w, float* __restrict__ acc_b)
{
    const int side = blockIdx.y;
    const float*  __restrict__ feats = side ? feats_b : feats_w;
    const __bf16* __restrict__ W     = side ? Wbbf : Wwbf;
    const float*  __restrict__ bias  = side ? bb : bw;
    float*        __restrict__ out   = side ? acc_b : acc_w;

    const int wave = threadIdx.x >> 5;
    const int lane = threadIdx.x & 31;
    const int l16  = lane & 15;
    const int h    = lane >> 4;          // lane half: 0 or 1

    const int rowBase = blockIdx.x * 32 + (wave >> 1) * 16;
    const int colBase = (wave & 1) * 64;       // four 16-wide N tiles per wave

    // A fragment: row = rowBase + l16; K runs at k0+16h .. +15 (contiguous).
    const float* __restrict__ aPtr = feats + (size_t)(rowBase + l16) * HD + 16 * h;
    // B fragments: lane needs W[col][k0 + 16h .. +15] = 16 contiguous bf16 (32B).
    const __bf16* bPtr[4];
#pragma unroll
    for (int t = 0; t < 4; ++t)
        bPtr[t] = W + (size_t)(colBase + t * 16 + l16) * HD + 16 * h;

    v8f c[4] = {{}, {}, {}, {}};

#pragma unroll 2
    for (int k0 = 0; k0 < HD; k0 += 32) {
        __builtin_prefetch(aPtr + k0 + 512, 0, 3);  // near-cache prefetch ahead

        // A: two contiguous 8-float runs -> 16 bf16 per lane
        v4f a0 = *(const v4f*)(aPtr + k0);
        v4f a1 = *(const v4f*)(aPtr + k0 + 4);
        v4f a2 = *(const v4f*)(aPtr + k0 + 8);
        v4f a3 = *(const v4f*)(aPtr + k0 + 12);
        v16bf af = cvt16(a0, a1, a2, a3);

#pragma unroll
        for (int t = 0; t < 4; ++t) {
            v16bf bf = *(const v16bf*)(bPtr[t] + k0);   // 2x global_load_b128
            c[t] = wmma_bf16(af, bf, c[t]);
        }
    }

    // Epilogue: C/D layout -> row = r + 8h, col = l16 (+16 per tile)
    const int orow = rowBase + 8 * h;
#pragma unroll
    for (int t = 0; t < 4; ++t) {
        const int col = colBase + t * 16 + l16;
        const float bia = bias[col];
#pragma unroll
        for (int r = 0; r < 8; ++r) {
            out[(size_t)(orow + r) * L1 + col] = fmaxf(c[t][r] + bia, 0.0f);
        }
    }
}

// ---------------------------------------------------------------------------
// Kernel 2: policy = acc @ W_p^T + b_p ; acc gathered via stm select.
// K = 256 (8 WMMA K-steps). Wave tile: 16 rows x 32 cols. W_p pre-cvt bf16.
// ---------------------------------------------------------------------------
__global__ __launch_bounds__(128) void policy_gemm(
    const float* __restrict__ wa, const float* __restrict__ ba,
    const unsigned char* __restrict__ stm,
    const __bf16* __restrict__ Wpbf, const float* __restrict__ bp,
    float* __restrict__ pol)
{
    const int wave = threadIdx.x >> 5;
    const int lane = threadIdx.x & 31;
    const int l16  = lane & 15;
    const int h    = lane >> 4;

    const int rowBase = blockIdx.x * 16;
    const int colBase = wave * 32;

    const int row = rowBase + l16;
    const bool s = stm[row] != 0;
    // acc[row, 0:128] = s ? white : black ; acc[row, 128:256] = other half
    const float* __restrict__ fr = (s ? wa : ba) + (size_t)row * L1;
    const float* __restrict__ sr = (s ? ba : wa) + (size_t)row * L1;

    const int col0 = colBase + l16;
    const __bf16* __restrict__ b0 = Wpbf + (size_t)col0 * (2 * L1) + 16 * h;
    const __bf16* __restrict__ b1 = Wpbf + (size_t)(col0 + 16) * (2 * L1) + 16 * h;

    v8f c0 = {}, c1 = {};

#pragma unroll
    for (int k0 = 0; k0 < 2 * L1; k0 += 32) {
        const int ka = k0 + 16 * h;       // 8-float run, never crosses 128
        const int kb = ka + 8;
        const float* pa = (ka < L1) ? (fr + ka) : (sr + ka - L1);
        const float* pb = (kb < L1) ? (fr + kb) : (sr + kb - L1);
        v4f a0 = *(const v4f*)(pa);
        v4f a1 = *(const v4f*)(pa + 4);
        v4f a2 = *(const v4f*)(pb);
        v4f a3 = *(const v4f*)(pb + 4);
        v16bf af = cvt16(a0, a1, a2, a3);

        c0 = wmma_bf16(af, *(const v16bf*)(b0 + k0), c0);
        c1 = wmma_bf16(af, *(const v16bf*)(b1 + k0), c1);
    }

    const int orow = rowBase + 8 * h;
    const float bia0 = bp[col0];
    const float bia1 = bp[col0 + 16];
#pragma unroll
    for (int r = 0; r < 8; ++r) {
        float* dst = pol + (size_t)(orow + r) * PO;
        dst[col0]      = c0[r] + bia0;
        dst[col0 + 16] = c1[r] + bia1;
    }
}

// ---------------------------------------------------------------------------
// Kernel 3: tiny value head. One thread per batch row.
// ---------------------------------------------------------------------------
__global__ __launch_bounds__(256) void value_head(
    const float* __restrict__ wa, const float* __restrict__ ba,
    const unsigned char* __restrict__ stm,
    const float* __restrict__ Wv1, const float* __restrict__ bv1,
    const float* __restrict__ Wv2, const float* __restrict__ bv2,
    float* __restrict__ quant)
{
    const int i = blockIdx.x * blockDim.x + threadIdx.x;
    if (i >= BATCH) return;
    const bool s = stm[i] != 0;
    const float* __restrict__ a0 = (s ? wa : ba) + (size_t)i * L1;
    const float* __restrict__ a1 = (s ? ba : wa) + (size_t)i * L1;

    float h[VH];
#pragma unroll
    for (int j = 0; j < VH; ++j) h[j] = bv1[j];

    for (int k = 0; k < L1; ++k) {
        const float v = a0[k];
#pragma unroll
        for (int j = 0; j < VH; ++j) h[j] = fmaf(v, Wv1[j * (2 * L1) + k], h[j]);
    }
    for (int k = 0; k < L1; ++k) {
        const float v = a1[k];
#pragma unroll
        for (int j = 0; j < VH; ++j) h[j] = fmaf(v, Wv1[j * (2 * L1) + L1 + k], h[j]);
    }
#pragma unroll
    for (int j = 0; j < VH; ++j) h[j] = fmaxf(h[j], 0.0f);

    float raw[NQ];
#pragma unroll
    for (int q = 0; q < NQ; ++q) {
        float t = bv2[q];
#pragma unroll
        for (int j = 0; j < VH; ++j) t = fmaf(h[j], Wv2[q * VH + j], t);
        raw[q] = t;
    }

    float run = raw[0];
    quant[(size_t)i * NQ] = run;
#pragma unroll
    for (int q = 1; q < NQ; ++q) {
        const float x = raw[q];
        run += fmaxf(x, 0.0f) + log1pf(expf(-fabsf(x)));  // stable softplus
        quant[(size_t)i * NQ + q] = run;
    }
}

// ---------------------------------------------------------------------------
extern "C" void kernel_launch(void* const* d_in, const int* in_sizes, int n_in,
                              void* d_out, int out_size, void* d_ws, size_t ws_size,
                              hipStream_t stream) {
    const float* wf  = (const float*)d_in[0];          // white_features
    const float* bfe = (const float*)d_in[1];          // black_features
    const unsigned char* stm = (const unsigned char*)d_in[2];  // bool (1B)
    const float* Ww  = (const float*)d_in[3];
    const float* bw  = (const float*)d_in[4];
    const float* Wb  = (const float*)d_in[5];
    const float* bb  = (const float*)d_in[6];
    const float* Wv1 = (const float*)d_in[7];
    const float* bv1 = (const float*)d_in[8];
    const float* Wv2 = (const float*)d_in[9];
    const float* bv2 = (const float*)d_in[10];
    const float* Wp  = (const float*)d_in[11];
    const float* bp  = (const float*)d_in[12];

    float* out   = (float*)d_out;
    float* quant = out;                       // BATCH * NQ
    float* pol   = out + (size_t)BATCH * NQ;  // BATCH * PO

    // Workspace layout (f32 accs first, then bf16 weight copies)
    float*  wa   = (float*)d_ws;                         // BATCH*L1 f32
    float*  ba   = wa + (size_t)BATCH * L1;              // BATCH*L1 f32
    __bf16* Wwbf = (__bf16*)(ba + (size_t)BATCH * L1);   // L1*HD bf16
    __bf16* Wbbf = Wwbf + (size_t)L1 * HD;               // L1*HD bf16
    __bf16* Wpbf = Wbbf + (size_t)L1 * HD;               // PO*2*L1 bf16

    const int nft = L1 * HD;          // 2,883,584 (mult of 8)
    const int np  = PO * 2 * L1;      // 32,768
    cvt_weights<<<(nft / 8 + 255) / 256, 256, 0, stream>>>(Ww, Wwbf, nft);
    cvt_weights<<<(nft / 8 + 255) / 256, 256, 0, stream>>>(Wb, Wbbf, nft);
    cvt_weights<<<(np  / 8 + 255) / 256, 256, 0, stream>>>(Wp, Wpbf, np);

    ft_gemm<<<dim3(BATCH / 32, 2), 128, 0, stream>>>(wf, bfe, Wwbf, bw, Wbbf, bb, wa, ba);
    policy_gemm<<<BATCH / 16, 128, 0, stream>>>(wa, ba, stm, Wpbf, bp, pol);
    value_head<<<BATCH / 256, 256, 0, stream>>>(wa, ba, stm, Wv1, bv1, Wv2, bv2, quant);
}
